// SS2D_Wrapper_40862318854619
// MI455X (gfx1250) — matrix operand order repaired and checked by reading, not compiled, not run on hardware
//
#include <hip/hip_runtime.h>
#include <stdint.h>

// Problem constants (B=1)
#define LQ    2304   // H*W
#define HH    48
#define WWD   48
#define CCH   96
#define DIM   192
#define NST   16
#define RR    6
#define KD    4
#define CK1   864    // 96*9 implicit-GEMM K for conv1
#define XROWS 38     // R + 2N rows of x_dbl
#define SCH   256    // selective-scan L-chunk staged in LDS

typedef __attribute__((ext_vector_type(16))) _Float16 v16h;
typedef __attribute__((ext_vector_type(8)))  float    v8f;
typedef __attribute__((ext_vector_type(4)))  unsigned int u32x4;
typedef __attribute__((ext_vector_type(8)))  int      i32x8;
typedef __attribute__((ext_vector_type(4)))  int      i32x4;

#ifndef __has_builtin
#define __has_builtin(x) 0
#endif
#if __has_builtin(__builtin_amdgcn_tensor_load_to_lds)
#define HAVE_TDM 1
#else
#define HAVE_TDM 0
#endif

// B-matrix (32x16 f16): lane half hf holds K = {hf*8 + 0..7} then {hf*8 + 16..23}.
__device__ __forceinline__ int b_k(int j, int hf) { return (j & 7) + (hf << 3) + ((j >> 3) << 4); }

// A-fragment loader: per ISA 7.12.2 the lane's 16 f16 elements are K =
// {hf*8 + 0..7} then {hf*8 + 16..23} -> two contiguous 8-float spans.
// p must point at row + k0 + hf*8 (16B aligned). Emits 4x global_load_b128.
__device__ __forceinline__ v16h load_a4(const float* __restrict__ p) {
  const float4* q = (const float4*)p;
  float4 q0 = q[0], q1 = q[1], q2 = q[4], q3 = q[5];  // +0,+4,+16,+20 floats
  v16h a;
  a[0]=(_Float16)q0.x; a[1]=(_Float16)q0.y; a[2]=(_Float16)q0.z; a[3]=(_Float16)q0.w;
  a[4]=(_Float16)q1.x; a[5]=(_Float16)q1.y; a[6]=(_Float16)q1.z; a[7]=(_Float16)q1.w;
  a[8]=(_Float16)q2.x; a[9]=(_Float16)q2.y; a[10]=(_Float16)q2.z; a[11]=(_Float16)q2.w;
  a[12]=(_Float16)q3.x; a[13]=(_Float16)q3.y; a[14]=(_Float16)q3.z; a[15]=(_Float16)q3.w;
  return a;
}
// B-fragment loader for K-contiguous operand: two 8-float spans at
// row + k0 + hf*8 and +16. Emits 4x global_load_b128.
__device__ __forceinline__ v16h load_b4(const float* __restrict__ p) {
  const float4* q = (const float4*)p;
  float4 q0 = q[0], q1 = q[1], q2 = q[4], q3 = q[5];
  v16h b;
  b[0]=(_Float16)q0.x; b[1]=(_Float16)q0.y; b[2]=(_Float16)q0.z; b[3]=(_Float16)q0.w;
  b[4]=(_Float16)q1.x; b[5]=(_Float16)q1.y; b[6]=(_Float16)q1.z; b[7]=(_Float16)q1.w;
  b[8]=(_Float16)q2.x; b[9]=(_Float16)q2.y; b[10]=(_Float16)q2.z; b[11]=(_Float16)q2.w;
  b[12]=(_Float16)q3.x; b[13]=(_Float16)q3.y; b[14]=(_Float16)q3.z; b[15]=(_Float16)q3.w;
  return b;
}

// ---------------------------------------------------------------------------
// K1: conv0 (1x1) as WMMA GEMM  out0[96,L] = W[96,96] * x[96,L], + BN + PReLU
// ---------------------------------------------------------------------------
__global__ __launch_bounds__(32)
void k_conv0(const float* __restrict__ x, const float* __restrict__ w,
             const float* __restrict__ bw, const float* __restrict__ bb,
             const float* __restrict__ bm, const float* __restrict__ bv,
             const float* __restrict__ pa, float* __restrict__ out) {
  const int tn = blockIdx.x % (LQ / 16);
  const int tm = blockIdx.x / (LQ / 16);
  const int lane = threadIdx.x;
  const int ml = lane & 15, hf = lane >> 4;
  const int m = tm * 16 + ml;
  const int n = tn * 16 + ml;
  v8f acc = {};
  for (int k0 = 0; k0 < CCH; k0 += 32) {
    v16h a = load_a4(w + m * CCH + k0 + hf * 8);
    v16h b;
#pragma unroll
    for (int j = 0; j < 16; ++j)
      b[j] = (_Float16)x[(size_t)(k0 + b_k(j, hf)) * LQ + n];
    acc = __builtin_amdgcn_wmma_f32_16x16x32_f16(false, a, false, b,
                                                 (short)0, acc, false, false);
  }
#pragma unroll
  for (int r = 0; r < 8; ++r) {
    int mo = tm * 16 + r + hf * 8;
    float s = bw[mo] * rsqrtf(bv[mo] + 1e-5f);
    float y = (acc[r] - bm[mo]) * s + bb[mo];
    out[mo * LQ + n] = y > 0.f ? y : pa[mo] * y;
  }
}

// ---------------------------------------------------------------------------
// K2: conv1 (3x3, pad 1) as implicit WMMA GEMM, K = 864, + BN + PReLU
// ---------------------------------------------------------------------------
__global__ __launch_bounds__(32)
void k_conv1(const float* __restrict__ in, const float* __restrict__ cw,
             const float* __restrict__ bw, const float* __restrict__ bb,
             const float* __restrict__ bm, const float* __restrict__ bv,
             const float* __restrict__ pa, float* __restrict__ out) {
  const int tn = blockIdx.x % (LQ / 16);
  const int tm = blockIdx.x / (LQ / 16);
  const int lane = threadIdx.x;
  const int ml = lane & 15, hf = lane >> 4;
  const int m = tm * 16 + ml;
  const int n = tn * 16 + ml;
  const int ph = n / WWD, pw = n % WWD;
  v8f acc = {};
  for (int k0 = 0; k0 < CK1; k0 += 32) {
    if (k0 + 32 < CK1)  // gfx1250 global_prefetch_b8 for next weight tile
      __builtin_prefetch((const void*)&cw[m * CK1 + k0 + 32], 0, 1);
    v16h a = load_a4(cw + m * CK1 + k0 + hf * 8);
    v16h b;
#pragma unroll
    for (int j = 0; j < 16; ++j) {
      int kb = k0 + b_k(j, hf);
      int c = kb / 9, rem = kb - c * 9;
      int hp = ph + rem / 3 - 1, wp = pw + rem % 3 - 1;
      float v = 0.f;
      if (hp >= 0 && hp < HH && wp >= 0 && wp < WWD)
        v = in[c * LQ + hp * WWD + wp];
      b[j] = (_Float16)v;
    }
    acc = __builtin_amdgcn_wmma_f32_16x16x32_f16(false, a, false, b,
                                                 (short)0, acc, false, false);
  }
#pragma unroll
  for (int r = 0; r < 8; ++r) {
    int mo = tm * 16 + r + hf * 8;
    float s = bw[mo] * rsqrtf(bv[mo] + 1e-5f);
    float y = (acc[r] - bm[mo]) * s + bb[mo];
    out[mo * LQ + n] = y > 0.f ? y : pa[mo] * y;
  }
}

// ---------------------------------------------------------------------------
// K3: in_proj WMMA GEMM  xz[384,L] = in_proj_w[384,96] * out1[96,L]
// ---------------------------------------------------------------------------
__global__ __launch_bounds__(32)
void k_inproj(const float* __restrict__ in, const float* __restrict__ w,
              float* __restrict__ xz) {
  const int tn = blockIdx.x % (LQ / 16);
  const int tm = blockIdx.x / (LQ / 16);
  const int lane = threadIdx.x;
  const int ml = lane & 15, hf = lane >> 4;
  const int m = tm * 16 + ml;
  const int n = tn * 16 + ml;
  v8f acc = {};
  for (int k0 = 0; k0 < CCH; k0 += 32) {
    v16h a = load_a4(w + m * CCH + k0 + hf * 8);
    v16h b;
#pragma unroll
    for (int j = 0; j < 16; ++j)
      b[j] = (_Float16)in[(size_t)(k0 + b_k(j, hf)) * LQ + n];
    acc = __builtin_amdgcn_wmma_f32_16x16x32_f16(false, a, false, b,
                                                 (short)0, acc, false, false);
  }
#pragma unroll
  for (int r = 0; r < 8; ++r)
    xz[(size_t)(tm * 16 + r + hf * 8) * LQ + n] = acc[r];
}

// ---------------------------------------------------------------------------
// K4: depthwise 3x3 conv + bias + SiLU; scatter into all 4 scan orders of xs
// ---------------------------------------------------------------------------
__global__ void k_dwconv(const float* __restrict__ xz, const float* __restrict__ w,
                         const float* __restrict__ bias, float* __restrict__ xs) {
  int idx = blockIdx.x * blockDim.x + threadIdx.x;
  if (idx >= DIM * LQ) return;
  int d = idx / LQ, l = idx % LQ;
  int h = l / WWD, pw = l % WWD;
  float acc = bias[d];
#pragma unroll
  for (int dy = -1; dy <= 1; ++dy)
#pragma unroll
    for (int dx = -1; dx <= 1; ++dx) {
      int hp = h + dy, wp = pw + dx;
      if (hp >= 0 && hp < HH && wp >= 0 && wp < WWD)
        acc += xz[d * LQ + hp * WWD + wp] * w[d * 9 + (dy + 1) * 3 + (dx + 1)];
    }
  float v = acc / (1.f + __expf(-acc));           // SiLU
  int lt = pw * HH + h;                           // H/W-transposed index
  xs[(size_t)(0 * DIM + d) * LQ + l] = v;
  xs[(size_t)(1 * DIM + d) * LQ + lt] = v;
  xs[(size_t)(2 * DIM + d) * LQ + (LQ - 1 - l)] = v;
  xs[(size_t)(3 * DIM + d) * LQ + (LQ - 1 - lt)] = v;
}

// ---------------------------------------------------------------------------
// K5: x_proj WMMA GEMM per direction: x_dbl[k][38,L] = x_proj_w[k][38,192]*xs[k]
//     (M padded 38 -> 48; pad rows zeroed and not stored)
// ---------------------------------------------------------------------------
__global__ __launch_bounds__(32)
void k_xproj(const float* __restrict__ xs, const float* __restrict__ w,
             float* __restrict__ xdbl) {
  const int tn = blockIdx.x;          // 144
  const int tm = blockIdx.y;          // 3 (48 rows)
  const int kdir = blockIdx.z;        // 4
  const int lane = threadIdx.x;
  const int ml = lane & 15, hf = lane >> 4;
  const int m = tm * 16 + ml;
  const int n = tn * 16 + ml;
  const float* wk = w + (size_t)kdir * XROWS * DIM;
  const float* xk = xs + (size_t)kdir * DIM * LQ;
  v8f acc = {};
  for (int k0 = 0; k0 < DIM; k0 += 32) {
    v16h a = {};
    if (m < XROWS) a = load_a4(wk + m * DIM + k0 + hf * 8);
    v16h b;
#pragma unroll
    for (int j = 0; j < 16; ++j)
      b[j] = (_Float16)xk[(size_t)(k0 + b_k(j, hf)) * LQ + n];
    acc = __builtin_amdgcn_wmma_f32_16x16x32_f16(false, a, false, b,
                                                 (short)0, acc, false, false);
  }
#pragma unroll
  for (int r = 0; r < 8; ++r) {
    int c = tm * 16 + r + hf * 8;
    if (c < XROWS) xdbl[(size_t)(kdir * XROWS + c) * LQ + n] = acc[r];
  }
}

// ---------------------------------------------------------------------------
// K6: dt projection (rank 6) + softplus
// ---------------------------------------------------------------------------
__global__ void k_dt(const float* __restrict__ xdbl, const float* __restrict__ dw,
                     const float* __restrict__ db, float* __restrict__ dts) {
  int idx = blockIdx.x * blockDim.x + threadIdx.x;
  if (idx >= KD * DIM * LQ) return;
  int l = idx % LQ, kd = idx / LQ;
  int kdir = kd / DIM;
  float acc = db[kd];
#pragma unroll
  for (int r = 0; r < RR; ++r)
    acc += dw[kd * RR + r] * xdbl[(size_t)(kdir * XROWS + r) * LQ + l];
  dts[idx] = acc > 20.f ? acc : log1pf(__expf(acc));
}

// ---------------------------------------------------------------------------
// K7: selective scan. One block per direction; thread d keeps 16 states in
//     VGPRs. The 32-row B/C panel (rows 6..37 of x_dbl, shared by all d) is
//     DMA'd to LDS in 256-column chunks by the Tensor Data Mover (wave 0
//     issues TENSOR_LOAD_TO_LDS, waits TENSORcnt, barrier releases block).
// ---------------------------------------------------------------------------
__global__ __launch_bounds__(DIM)
void k_scan(const float* __restrict__ xs, const float* __restrict__ dts,
            const float* __restrict__ xdbl, const float* __restrict__ alog,
            const float* __restrict__ Ds, float* __restrict__ oy) {
  const int kdir = blockIdx.x;
  const int d = threadIdx.x;
  __shared__ float sPanel[32 * SCH];   // rows 0..15 = B(n), rows 16..31 = C(n)
  float A[NST], hst[NST];
#pragma unroll
  for (int nn = 0; nn < NST; ++nn) {
    A[nn] = -__expf(alog[(kdir * DIM + d) * NST + nn]);
    hst[nn] = 0.f;
  }
  const float Dp = Ds[kdir * DIM + d];
  const float* xrow  = xs  + (size_t)(kdir * DIM + d) * LQ;
  const float* dtrow = dts + (size_t)(kdir * DIM + d) * LQ;
  const float* panel = xdbl + (size_t)(kdir * XROWS + RR) * LQ;  // 32 rows
  float* orow = oy + (size_t)(kdir * DIM + d) * LQ;

  for (int ch = 0; ch < LQ / SCH; ++ch) {
#if HAVE_TDM
    if (threadIdx.x < 32) {   // wave 0 only issues the TDM descriptor
      unsigned long long ga = (unsigned long long)(uintptr_t)(panel + ch * SCH);
      u32x4 g0;
      g0[0] = 1u;                                        // count=1, user mode
      g0[1] = (unsigned)(uintptr_t)&sPanel[0];           // LDS byte address
      g0[2] = (unsigned)ga;                              // global_addr[31:0]
      g0[3] = (unsigned)(ga >> 32) | (2u << 30);         // addr[56:32] | type=2
      i32x8 g1;
      g1[0] = 0x00020000;                                // data_size = 4B
      g1[1] = (int)(((unsigned)LQ & 0xffffu) << 16);     // tensor_dim0 lo16
      g1[2] = (int)(((unsigned)LQ >> 16) | (32u << 16)); // dim0 hi | dim1 lo
      g1[3] = (int)((unsigned)SCH << 16);                // dim1 hi=0 | tile_dim0
      g1[4] = 32;                                        // tile_dim1 (tile_dim2=0)
      g1[5] = LQ;                                        // dim0_stride lo32
      g1[6] = (int)(((unsigned)LQ & 0xffffu) << 16);     // stride0 hi=0|stride1 lo
      g1[7] = (int)((unsigned)LQ >> 16);                 // dim1_stride hi32
      i32x4 z4 = {};
#if __clang_major__ >= 23
      i32x8 z8 = {};
      __builtin_amdgcn_tensor_load_to_lds(g0, g1, z4, z4, z8, 0);
#else
      __builtin_amdgcn_tensor_load_to_lds(g0, g1, z4, z4, 0);
#endif
#if __has_builtin(__builtin_amdgcn_s_wait_tensorcnt)
      __builtin_amdgcn_s_wait_tensorcnt(0);
#else
      asm volatile("s_wait_tensorcnt 0x0" ::: "memory");
#endif
    }
#else
    for (int idx = threadIdx.x; idx < 32 * SCH; idx += DIM)
      sPanel[idx] = panel[(size_t)(idx / SCH) * LQ + ch * SCH + (idx % SCH)];
#endif
    __syncthreads();
    for (int i = 0; i < SCH; ++i) {
      int l = ch * SCH + i;
      float dt = dtrow[l], xv = xrow[l];
      float dBx = dt * xv;
      float y = 0.f;
#pragma unroll
      for (int nn = 0; nn < NST; ++nn) {
        float dA = __expf(dt * A[nn]);
        hst[nn] = dA * hst[nn] + dBx * sPanel[nn * SCH + i];
        y += sPanel[(NST + nn) * SCH + i] * hst[nn];
      }
      orow[l] = y + Dp * xv;
    }
    __syncthreads();
  }
}

// ---------------------------------------------------------------------------
// K8: combine 4 directions + LayerNorm(DI) + SiLU(z) gate -> ycomb[L,DI]
// ---------------------------------------------------------------------------
__global__ __launch_bounds__(DIM)
void k_combine(const float* __restrict__ oy, const float* __restrict__ xz,
               const float* __restrict__ nw, const float* __restrict__ nb,
               float* __restrict__ yc) {
  const int l = blockIdx.x;
  const int d = threadIdx.x;
  const int h = l / WWD, pw = l % WWD;
  const int lt = pw * HH + h;
  float v = oy[(size_t)(0 * DIM + d) * LQ + l]
          + oy[(size_t)(2 * DIM + d) * LQ + (LQ - 1 - l)]
          + oy[(size_t)(1 * DIM + d) * LQ + lt]
          + oy[(size_t)(3 * DIM + d) * LQ + (LQ - 1 - lt)];
  __shared__ float red[DIM];
  red[d] = v;
  __syncthreads();
  if (d < 64) red[d] += red[d + 64] + red[d + 128];
  __syncthreads();
  for (int s = 32; s >= 1; s >>= 1) {
    if (d < s) red[d] += red[d + s];
    __syncthreads();
  }
  float mu = red[0] * (1.f / DIM);
  __syncthreads();
  float c = v - mu;
  red[d] = c * c;
  __syncthreads();
  if (d < 64) red[d] += red[d + 64] + red[d + 128];
  __syncthreads();
  for (int s = 32; s >= 1; s >>= 1) {
    if (d < s) red[d] += red[d + s];
    __syncthreads();
  }
  float var = red[0] * (1.f / DIM);
  float yn = c * rsqrtf(var + 1e-5f) * nw[d] + nb[d];
  float z = xz[(size_t)(DIM + d) * LQ + l];
  yc[(size_t)l * DIM + d] = yn * (z / (1.f + __expf(-z)));
}

// ---------------------------------------------------------------------------
// K9: out_proj WMMA GEMM + gamma * ss + residual  (both operands K-contiguous)
// ---------------------------------------------------------------------------
__global__ __launch_bounds__(32)
void k_outproj(const float* __restrict__ yc, const float* __restrict__ w,
               const float* __restrict__ res, const float* __restrict__ gamma,
               float* __restrict__ out) {
  const int tn = blockIdx.x % (LQ / 16);
  const int tm = blockIdx.x / (LQ / 16);
  const int lane = threadIdx.x;
  const int ml = lane & 15, hf = lane >> 4;
  const int m = tm * 16 + ml;
  const int n = tn * 16 + ml;
  const float g = gamma[0];
  v8f acc = {};
  for (int k0 = 0; k0 < DIM; k0 += 32) {
    v16h a = load_a4(w + m * DIM + k0 + hf * 8);
    v16h b = load_b4(yc + (size_t)n * DIM + k0 + hf * 8);
    acc = __builtin_amdgcn_wmma_f32_16x16x32_f16(false, a, false, b,
                                                 (short)0, acc, false, false);
  }
#pragma unroll
  for (int r = 0; r < 8; ++r) {
    int mo = tm * 16 + r + hf * 8;
    out[mo * LQ + n] = g * acc[r] + res[mo * LQ + n];
  }
}

// ---------------------------------------------------------------------------
extern "C" void kernel_launch(void* const* d_in, const int* in_sizes, int n_in,
                              void* d_out, int out_size, void* d_ws, size_t ws_size,
                              hipStream_t stream) {
  const float* x        = (const float*)d_in[0];
  const float* conv0_w  = (const float*)d_in[1];
  const float* bn0_w    = (const float*)d_in[2];
  const float* bn0_b    = (const float*)d_in[3];
  const float* bn0_m    = (const float*)d_in[4];
  const float* bn0_v    = (const float*)d_in[5];
  const float* prelu0   = (const float*)d_in[6];
  const float* conv1_w  = (const float*)d_in[7];
  const float* bn1_w    = (const float*)d_in[8];
  const float* bn1_b    = (const float*)d_in[9];
  const float* bn1_m    = (const float*)d_in[10];
  const float* bn1_v    = (const float*)d_in[11];
  const float* prelu1   = (const float*)d_in[12];
  const float* in_proj  = (const float*)d_in[13];
  const float* conv2d_w = (const float*)d_in[14];
  const float* conv2d_b = (const float*)d_in[15];
  const float* x_proj_w = (const float*)d_in[16];
  const float* dt_w     = (const float*)d_in[17];
  const float* dt_b     = (const float*)d_in[18];
  const float* A_logs   = (const float*)d_in[19];
  const float* Ds       = (const float*)d_in[20];
  const float* norm_w   = (const float*)d_in[21];
  const float* norm_b   = (const float*)d_in[22];
  const float* outp_w   = (const float*)d_in[23];
  const float* gamma    = (const float*)d_in[24];

  float* ws   = (float*)d_ws;
  float* out0 = ws;                       // 96*2304
  float* out1 = out0 + CCH * LQ;          // 96*2304 (residual)
  float* xz   = out1 + CCH * LQ;          // 384*2304
  float* xs   = xz + 2 * DIM * LQ;        // 4*192*2304
  float* xdbl = xs + KD * DIM * LQ;       // 4*38*2304
  float* dts  = xdbl + KD * XROWS * LQ;   // 4*192*2304
  float* oy   = dts + KD * DIM * LQ;      // 4*192*2304
  float* yc   = oy + KD * DIM * LQ;       // 2304*192

  k_conv0<<<dim3((CCH / 16) * (LQ / 16)), dim3(32), 0, stream>>>(
      x, conv0_w, bn0_w, bn0_b, bn0_m, bn0_v, prelu0, out0);
  k_conv1<<<dim3((CCH / 16) * (LQ / 16)), dim3(32), 0, stream>>>(
      out0, conv1_w, bn1_w, bn1_b, bn1_m, bn1_v, prelu1, out1);
  k_inproj<<<dim3((2 * DIM / 16) * (LQ / 16)), dim3(32), 0, stream>>>(
      out1, in_proj, xz);
  k_dwconv<<<dim3((DIM * LQ + 255) / 256), dim3(256), 0, stream>>>(
      xz, conv2d_w, conv2d_b, xs);
  k_xproj<<<dim3(LQ / 16, 3, KD), dim3(32), 0, stream>>>(xs, x_proj_w, xdbl);
  k_dt<<<dim3((KD * DIM * LQ + 255) / 256), dim3(256), 0, stream>>>(
      xdbl, dt_w, dt_b, dts);
  k_scan<<<dim3(KD), dim3(DIM), 0, stream>>>(xs, dts, xdbl, A_logs, Ds, oy);
  k_combine<<<dim3(LQ), dim3(DIM), 0, stream>>>(oy, xz, norm_w, norm_b, yc);
  k_outproj<<<dim3((CCH / 16) * (LQ / 16)), dim3(32), 0, stream>>>(
      yc, outp_w, out1, gamma, (float*)d_out);
}
